// ConvNeXtEncoderBlock_70557722738748
// MI455X (gfx1250) — matrix operand order repaired and checked by reading, not compiled, not run on hardware
//
#include <hip/hip_runtime.h>
#include <math.h>

#define BB 16
#define CC 96
#define HH 56
#define WWW 56
#define KK 7
#define PP (HH * WWW)   // 3136
#define C4 (4 * CC)     // 384

typedef __attribute__((ext_vector_type(16))) __bf16 v16bf;
typedef __attribute__((ext_vector_type(8)))  float  v8f;

// ---------------------------------------------------------------------------
// depthwise 7x7 conv, padding SAME, + bias; interior fast path (no bounds checks)
// ---------------------------------------------------------------------------
__global__ void dw_conv_kernel(const float* __restrict__ x,
                               const float* __restrict__ w,
                               const float* __restrict__ bias,
                               float* __restrict__ out) {
    int idx = blockIdx.x * blockDim.x + threadIdx.x;
    if (idx >= BB * CC * PP) return;
    int p = idx % PP;
    int c = (idx / PP) % CC;
    int b = idx / (CC * PP);
    int yy = p / WWW, xx = p % WWW;
    const float* xp = x + ((size_t)b * CC + c) * PP;
    const float* wp = w + c * KK * KK;
    float acc = bias[c];
    if (yy >= 3 && yy < HH - 3 && xx >= 3 && xx < WWW - 3) {
        const float* xc = xp + (yy - 3) * WWW + (xx - 3);
        #pragma unroll
        for (int dy = 0; dy < KK; ++dy)
            #pragma unroll
            for (int dx = 0; dx < KK; ++dx)
                acc += xc[dy * WWW + dx] * wp[dy * KK + dx];
    } else {
        #pragma unroll
        for (int dy = 0; dy < KK; ++dy) {
            int sy = yy + dy - 3;
            if (sy < 0 || sy >= HH) continue;
            #pragma unroll
            for (int dx = 0; dx < KK; ++dx) {
                int sx = xx + dx - 3;
                if (sx < 0 || sx >= WWW) continue;
                acc += xp[sy * WWW + sx] * wp[dy * KK + dx];
            }
        }
    }
    out[idx] = acc;
}

// ---------------------------------------------------------------------------
// Register-blocked WMMA bf16 16x16x32 core: one wave computes MT 16x16 tiles
// stacked along D, sharing a single B fragment per K-step.  K fully unrolled.
// Fragment layouts used here (wave32; A/B use contiguous K, not the ISA table order):
//   A 16x32 bf16: half h=lane>>4, row m=lane&15, elem i -> K = 16h + i
//   B 32x16 bf16: col n=lane&15, elem i -> K = 16h + i
//   C/D f32: vgpr r: M = r + 8h, N = lane&15
// Optional folds on the B element:
//   LN : v = (v - mu)*rs*lw[c,p] + lb[c,p]   (same (c,p) offset as the element)
//   GRN: v = v*mult[c] + addv[c]
// ---------------------------------------------------------------------------
template <int MT, int KDIM, bool LN, bool GRN>
__device__ __forceinline__ void gemm_core(const float* __restrict__ A,    // W + d0*KDIM
                                          const float* __restrict__ Bm,   // act + b-off + p0
                                          const float* __restrict__ lwp,  // ln_w + p0
                                          const float* __restrict__ lbp,  // ln_b + p0
                                          float mu, float rs,
                                          const float* __restrict__ mult, // + b*KDIM
                                          const float* __restrict__ addv,
                                          v8f acc[MT]) {
    const int lane = threadIdx.x & 31;
    const int half = lane >> 4;
    const int m    = lane & 15;
    #pragma unroll
    for (int k0 = 0; k0 < KDIM; k0 += 32) {
        if (k0 + 32 < KDIM)  // static after unroll -> clean global_prefetch_b8
            __builtin_prefetch(Bm + (size_t)(k0 + 32 + half * 16) * PP + m, 0, 3);
        v16bf bfrag;
        #pragma unroll
        for (int i = 0; i < 16; ++i) {
            const int kb = k0 + half * 16 + i;
            float v = Bm[(size_t)kb * PP + m];
            if (LN)  v = (v - mu) * rs * lwp[(size_t)kb * PP + m] + lbp[(size_t)kb * PP + m];
            if (GRN) v = v * mult[kb] + addv[kb];
            bfrag[i] = (__bf16)v;
        }
        #pragma unroll
        for (int t = 0; t < MT; ++t) {
            v16bf a;
            #pragma unroll
            for (int i = 0; i < 16; ++i) {
                const int ka = half * 16 + i;
                a[i] = (__bf16)A[(t * 16 + m) * KDIM + k0 + ka];
            }
            acc[t] = __builtin_amdgcn_wmma_f32_16x16x32_bf16(
                false, a, false, bfrag, (short)0, acc[t], false, false);
        }
    }
}

__device__ __forceinline__ float gelu_tanh_f(float x) {
    const float c = 0.7978845608028654f;  // sqrt(2/pi)
    float x3 = x * x * x;
    return 0.5f * x * (1.0f + tanhf(c * (x + 0.044715f * x3)));
}

// ---------------------------------------------------------------------------
// pointwise 1x1 (D=96, K=96): 8 waves = 2 D-groups (MT=3 -> 48 rows) x 4 P-waves
// grid = (1, 49, B), block = 256
// ---------------------------------------------------------------------------
__global__ void __launch_bounds__(256) pw_gemm_kernel(const float* __restrict__ act,
                                                      const float* __restrict__ Wt,
                                                      const float* __restrict__ bias,
                                                      float* __restrict__ out) {
    const int w  = threadIdx.x >> 5;
    const int wd = w & 1, wp = w >> 1;
    const int lane = threadIdx.x & 31;
    const int d0 = wd * 48;
    const int p0 = (blockIdx.y * 4 + wp) * 16;
    const int b  = blockIdx.z;
    v8f acc[3] = {};
    gemm_core<3, CC, false, false>(Wt + d0 * CC, act + (size_t)b * CC * PP + p0,
                                   nullptr, nullptr, 0.f, 0.f, nullptr, nullptr, acc);
    const int half = lane >> 4, n = lane & 15;
    #pragma unroll
    for (int t = 0; t < 3; ++t) {
        #pragma unroll
        for (int r = 0; r < 8; ++r) {
            int d = d0 + t * 16 + half * 8 + r;
            out[((size_t)b * CC + d) * PP + p0 + n] = acc[t][r] + bias[d];
        }
    }
}

// expand 1x1 (D=384, K=96) with LayerNorm folded into the B load; + bias + GELU
// grid = (3, 49, B): d0 = (bx*2+wd)*64 (MT=4)
__global__ void __launch_bounds__(256) ex_gemm_kernel(const float* __restrict__ h1,
                                                      const float* __restrict__ stats,
                                                      const float* __restrict__ ln_w,
                                                      const float* __restrict__ ln_b,
                                                      const float* __restrict__ Wt,
                                                      const float* __restrict__ bias,
                                                      float* __restrict__ pre) {
    const int w  = threadIdx.x >> 5;
    const int wd = w & 1, wp = w >> 1;
    const int lane = threadIdx.x & 31;
    const int d0 = (blockIdx.x * 2 + wd) * 64;
    const int p0 = (blockIdx.y * 4 + wp) * 16;
    const int b  = blockIdx.z;
    const float inv = 1.0f / (float)(CC * PP);
    const float mu  = stats[b * 2 + 0] * inv;
    const float var = stats[b * 2 + 1] * inv - mu * mu;
    const float rs  = rsqrtf(var + 1e-6f);
    v8f acc[4] = {};
    gemm_core<4, CC, true, false>(Wt + d0 * CC, h1 + (size_t)b * CC * PP + p0,
                                  ln_w + p0, ln_b + p0, mu, rs, nullptr, nullptr, acc);
    const int half = lane >> 4, n = lane & 15;
    #pragma unroll
    for (int t = 0; t < 4; ++t) {
        #pragma unroll
        for (int r = 0; r < 8; ++r) {
            int d = d0 + t * 16 + half * 8 + r;
            float v = acc[t][r] + bias[d];
            pre[((size_t)b * C4 + d) * PP + p0 + n] = gelu_tanh_f(v);
        }
    }
}

// projection 1x1 (D=96, K=384) with GRN affine folded into the B load;
// epilogue + l2_b + residual x -> final output.  grid = (1, 49, B)
__global__ void __launch_bounds__(256) proj_gemm_kernel(const float* __restrict__ pre,
                                                        const float* __restrict__ Wt,
                                                        const float* __restrict__ bias,
                                                        const float* __restrict__ mult,
                                                        const float* __restrict__ addv,
                                                        const float* __restrict__ x,
                                                        float* __restrict__ out) {
    const int w  = threadIdx.x >> 5;
    const int wd = w & 1, wp = w >> 1;
    const int lane = threadIdx.x & 31;
    const int d0 = wd * 48;
    const int p0 = (blockIdx.y * 4 + wp) * 16;
    const int b  = blockIdx.z;
    v8f acc[3] = {};
    gemm_core<3, C4, false, true>(Wt + d0 * C4, pre + (size_t)b * C4 * PP + p0,
                                  nullptr, nullptr, 0.f, 0.f,
                                  mult + (size_t)b * C4, addv + (size_t)b * C4, acc);
    const int half = lane >> 4, n = lane & 15;
    #pragma unroll
    for (int t = 0; t < 3; ++t) {
        #pragma unroll
        for (int r = 0; r < 8; ++r) {
            int d = d0 + t * 16 + half * 8 + r;
            size_t oi = ((size_t)b * CC + d) * PP + p0 + n;
            out[oi] = acc[t][r] + bias[d] + x[oi];
        }
    }
}

// ---------------------------------------------------------------------------
// LayerNorm statistics over (C,H,W) per sample
// ---------------------------------------------------------------------------
__global__ void zero_stats_kernel(float* __restrict__ stats) {
    if (threadIdx.x < 2 * BB) stats[threadIdx.x] = 0.0f;
}

__global__ void ln_stats_kernel(const float* __restrict__ h, float* __restrict__ stats) {
    const int b = blockIdx.x;
    __shared__ float ssum[256], ssq[256];
    const float* hb = h + (size_t)b * CC * PP;
    float s = 0.f, q = 0.f;
    const int base = blockIdx.y * 2048 + threadIdx.x;
    #pragma unroll
    for (int i = 0; i < 8; ++i) {
        float v = hb[base + i * 256];
        s += v; q += v * v;
    }
    ssum[threadIdx.x] = s; ssq[threadIdx.x] = q;
    __syncthreads();
    for (int off = 128; off > 0; off >>= 1) {
        if (threadIdx.x < off) {
            ssum[threadIdx.x] += ssum[threadIdx.x + off];
            ssq[threadIdx.x]  += ssq[threadIdx.x + off];
        }
        __syncthreads();
    }
    if (threadIdx.x == 0) {
        atomicAdd(&stats[b * 2 + 0], ssum[0]);
        atomicAdd(&stats[b * 2 + 1], ssq[0]);
    }
}

// ---------------------------------------------------------------------------
// largest singular value of each 56x56 slice via power iteration (LDS-resident)
// ---------------------------------------------------------------------------
__global__ void power_iter_kernel(const float* __restrict__ pre, float* __restrict__ sv) {
    __shared__ float M[HH * WWW];
    __shared__ float vvec[WWW], uvec[HH];
    __shared__ float nrm, sig;
    const float* src = pre + (size_t)blockIdx.x * PP;
    for (int i = threadIdx.x; i < PP; i += blockDim.x) M[i] = src[i];
    if (threadIdx.x < WWW) vvec[threadIdx.x] = 1.0f;
    __syncthreads();
    for (int it = 0; it < 12; ++it) {
        if (threadIdx.x < HH) {                    // u = M v
            float a = 0.f;
            #pragma unroll 8
            for (int c2 = 0; c2 < WWW; ++c2) a += M[threadIdx.x * WWW + c2] * vvec[c2];
            uvec[threadIdx.x] = a;
        }
        __syncthreads();
        if (threadIdx.x == 0) {
            float s = 0.f;
            for (int r = 0; r < HH; ++r) s += uvec[r] * uvec[r];
            nrm = rsqrtf(s + 1e-30f);
        }
        __syncthreads();
        if (threadIdx.x < HH) uvec[threadIdx.x] *= nrm;
        __syncthreads();
        if (threadIdx.x < WWW) {                   // v = M^T u
            float a = 0.f;
            #pragma unroll 8
            for (int r = 0; r < HH; ++r) a += M[r * WWW + threadIdx.x] * uvec[r];
            vvec[threadIdx.x] = a;
        }
        __syncthreads();
        if (threadIdx.x == 0) {
            float s = 0.f;
            for (int c2 = 0; c2 < WWW; ++c2) s += vvec[c2] * vvec[c2];
            s = sqrtf(s);
            sig = s;
            nrm = (s > 0.f) ? (1.0f / s) : 0.f;
        }
        __syncthreads();
        if (threadIdx.x < WWW) vvec[threadIdx.x] *= nrm;
        __syncthreads();
    }
    if (threadIdx.x == 0) sv[blockIdx.x] = sig;
}

__global__ void sv_reduce_kernel(const float* __restrict__ sv, float* __restrict__ svsum) {
    __shared__ float s[256];
    float a = 0.f;
    for (int i = threadIdx.x; i < BB * C4; i += 256) a += sv[i];
    s[threadIdx.x] = a;
    __syncthreads();
    for (int off = 128; off > 0; off >>= 1) {
        if (threadIdx.x < off) s[threadIdx.x] += s[threadIdx.x + off];
        __syncthreads();
    }
    if (threadIdx.x == 0) svsum[0] = s[0];
}

// h = gamma*(pre*svn) + beta + pre  ==  pre*(gamma*svn + 1) + beta
__global__ void grn_coef_kernel(const float* __restrict__ sv,
                                const float* __restrict__ svsum,
                                const float* __restrict__ gamma,
                                const float* __restrict__ beta,
                                float* __restrict__ mult,
                                float* __restrict__ addv) {
    int i = blockIdx.x * blockDim.x + threadIdx.x;
    if (i >= BB * C4) return;
    float svn = sv[i] / svsum[0];
    mult[i] = gamma[i] * svn + 1.0f;
    addv[i] = beta[i];
}

// ---------------------------------------------------------------------------
extern "C" void kernel_launch(void* const* d_in, const int* in_sizes, int n_in,
                              void* d_out, int out_size, void* d_ws, size_t ws_size,
                              hipStream_t stream) {
    const float* x     = (const float*)d_in[0];
    const float* dw_w  = (const float*)d_in[1];
    const float* dw_b  = (const float*)d_in[2];
    const float* pw_w  = (const float*)d_in[3];
    const float* pw_b  = (const float*)d_in[4];
    const float* ln_w  = (const float*)d_in[5];
    const float* ln_b  = (const float*)d_in[6];
    const float* l1_w  = (const float*)d_in[7];
    const float* l1_b  = (const float*)d_in[8];
    const float* gamma = (const float*)d_in[9];
    const float* beta  = (const float*)d_in[10];
    const float* l2_w  = (const float*)d_in[11];
    const float* l2_b  = (const float*)d_in[12];
    float* out = (float*)d_out;

    float* ws = (float*)d_ws;
    size_t o = 0;
    float* hd    = ws + o; o += (size_t)BB * CC * PP;   // depthwise output
    float* h1    = ws + o; o += (size_t)BB * CC * PP;   // pointwise output
    float* pre   = ws + o; o += (size_t)BB * C4 * PP;   // expand+GELU output
    float* sv    = ws + o; o += (size_t)BB * C4;
    float* stats = ws + o; o += 2 * BB;
    float* svsum = ws + o; o += 1;
    float* mult  = ws + o; o += (size_t)BB * C4;
    float* addv  = ws + o; o += (size_t)BB * C4;
    (void)ws_size; (void)in_sizes; (void)n_in; (void)out_size;

    const int nElemC = BB * CC * PP;

    dw_conv_kernel<<<(nElemC + 255) / 256, 256, 0, stream>>>(x, dw_w, dw_b, hd);

    pw_gemm_kernel<<<dim3(1, PP / 64, BB), 256, 0, stream>>>(hd, pw_w, pw_b, h1);

    zero_stats_kernel<<<1, 64, 0, stream>>>(stats);
    ln_stats_kernel<<<dim3(BB, (CC * PP) / 2048), 256, 0, stream>>>(h1, stats);

    // LayerNorm folded into the expand GEMM's B-operand load
    ex_gemm_kernel<<<dim3(3, PP / 64, BB), 256, 0, stream>>>(h1, stats, ln_w, ln_b,
                                                             l1_w, l1_b, pre);

    power_iter_kernel<<<BB * C4, 64, 0, stream>>>(pre, sv);
    sv_reduce_kernel<<<1, 256, 0, stream>>>(sv, svsum);
    grn_coef_kernel<<<(BB * C4 + 255) / 256, 256, 0, stream>>>(sv, svsum, gamma, beta, mult, addv);

    proj_gemm_kernel<<<dim3(1, PP / 64, BB), 256, 0, stream>>>(pre, l2_w, l2_b, mult, addv, x, out);
}
